// Mamba2Layer_23562190586217
// MI455X (gfx1250) — matrix operand order repaired and checked by reading, not compiled, not run on hardware
//
#include <hip/hip_runtime.h>
#include <hip/hip_bf16.h>

// ---------------------------------------------------------------------------
// Mamba2 layer forward for MI455X (gfx1250), wave32 + WMMA bf16 (fp32 accum).
// Pipeline:
//   1) rmsnorm(x)*w            -> h (bf16)
//   2) h @ w_in[:, :2304]      -> zxbcdt (WMMA bf16 GEMM)   [z | xBC]
//      h @ w_in[:, 2304:2308]  -> dt_raw (tiny dot kernel)
//   3) causal depthwise conv(K=4) + bias + SiLU over xBC -> xconv
//   4) chunked SSD scan (chunk L=64) with state in LDS, all matmuls via WMMA
//   5) rmsnorm(y * silu(z)) * gw -> ygn (bf16)
//   6) out = x + ygn @ w_out   (WMMA bf16 GEMM + residual epilogue)
// GEMM tiles staged with GLOBAL_LOAD_ASYNC_TO_LDS (ASYNCcnt-tracked DMA).
// ---------------------------------------------------------------------------

typedef __bf16 bf16;
typedef __attribute__((ext_vector_type(16))) __bf16 v16bf;
typedef __attribute__((ext_vector_type(8)))  float  v8f;

#define HID 512
#define INTER 1024
#define NH 4
#define PDIM 256
#define NDIM 128
#define KCONV 4
#define CONVD (INTER + 2 * NDIM)      // 1280
#define PROJ_W 2304                   // z + xBC columns (WMMA part)
#define PROJ_FULL 2308
#define BATCH 4
#define TLEN 4096
#define ROWS (BATCH * TLEN)           // 16384
#define EPSN 1e-5f
#define CHUNK 64
#define PT 32                         // P-tile per scan block

// ---------------- CDNA5 async global->LDS staging (guarded) ----------------

#if defined(__has_builtin)
#if __has_builtin(__builtin_amdgcn_global_load_async_to_lds_b128)
#define USE_ASYNC_LDS 1
#endif
#if __has_builtin(__builtin_amdgcn_s_wait_asynccnt)
#define WAIT_ASYNC() __builtin_amdgcn_s_wait_asynccnt(0)
#endif
#endif
#ifndef WAIT_ASYNC
#define WAIT_ASYNC() asm volatile("s_wait_asynccnt 0" ::: "memory")
#endif

#ifdef USE_ASYNC_LDS
typedef int v4i_t __attribute__((ext_vector_type(4)));
typedef __attribute__((address_space(1))) v4i_t* gasync_ptr;  // global src
typedef __attribute__((address_space(3))) v4i_t* lasync_ptr;  // LDS dst
__device__ __forceinline__ void async_cp16(void* lds, const void* gsrc) {
  // per-lane 16B DMA: LDS[lds] = MEM[gsrc]; tracked by ASYNCcnt
  __builtin_amdgcn_global_load_async_to_lds_b128((gasync_ptr)gsrc,
                                                 (lasync_ptr)lds, 0, 0);
}
#endif

// ---------------- WMMA helpers (CDNA5 16x16x32 bf16, wave32) ----------------

__device__ __forceinline__ v8f zero8() {
  v8f z = {0.f, 0.f, 0.f, 0.f, 0.f, 0.f, 0.f, 0.f};
  return z;
}

__device__ __forceinline__ v8f wmma_bf16(v16bf a, v16bf b, v8f c) {
  return __builtin_amdgcn_wmma_f32_16x16x32_bf16(
      false, a, false, b, (short)0, c, false, false);
}

// A (16x32, bf16): lane m = lane&15, elem j -> k = (lane>>4)*16 + j
__device__ __forceinline__ v16bf frag_a_bf(const bf16* base, int ld, int m0,
                                           int k0, int lane) {
  const bf16* p = base + (size_t)(m0 + (lane & 15)) * ld + k0 + ((lane >> 4) << 4);
  v16bf a;
#pragma unroll
  for (int j = 0; j < 8; ++j) a[j] = p[j];
#pragma unroll
  for (int j = 0; j < 8; ++j) a[j + 8] = p[j + 8];
  return a;
}

// B (32x16, bf16): lane n = lane&15, elem j -> k = (lane>>4)*16 + j
__device__ __forceinline__ v16bf frag_b_bf(const bf16* base, int ld, int k0,
                                           int n0, int lane) {
  const bf16* p = base + (size_t)(k0 + ((lane >> 4) << 4)) * ld + n0 + (lane & 15);
  v16bf b;
#pragma unroll
  for (int j = 0; j < 16; ++j) b[j] = p[(size_t)j * ld];
  return b;
}

// B operand sourced from fp32 LDS, converted on the fly
__device__ __forceinline__ v16bf frag_b_f32(const float* base, int ld, int k0,
                                            int n0, int lane) {
  const float* p = base + (size_t)(k0 + ((lane >> 4) << 4)) * ld + n0 + (lane & 15);
  v16bf b;
#pragma unroll
  for (int j = 0; j < 16; ++j) b[j] = (bf16)p[(size_t)j * ld];
  return b;
}

// B operand whose source matrix is stored transposed: value = base[col][k]
__device__ __forceinline__ v16bf frag_bT_bf(const bf16* base, int ld, int k0,
                                            int col0, int lane) {
  const bf16* p = base + (size_t)(col0 + (lane & 15)) * ld + k0 + ((lane >> 4) << 4);
  v16bf b;
#pragma unroll
  for (int j = 0; j < 16; ++j) b[j] = p[j];
  return b;
}

// ---------------- small elementwise / reduction kernels ----------------

__global__ __launch_bounds__(256) void cvt_w_in_kernel(
    const float* __restrict__ w, bf16* __restrict__ o) {
  int idx = blockIdx.x * 256 + threadIdx.x;  // 512*2304 total
  int r = idx / PROJ_W, c = idx - r * PROJ_W;
  o[idx] = (bf16)w[(size_t)r * PROJ_FULL + c];
}

__global__ __launch_bounds__(256) void cvt_w_out_kernel(
    const float* __restrict__ w, bf16* __restrict__ o) {
  int idx = blockIdx.x * 256 + threadIdx.x;  // 1024*512 total
  o[idx] = (bf16)w[idx];
}

__global__ __launch_bounds__(256) void rmsnorm_in_kernel(
    const float* __restrict__ x, const float* __restrict__ w,
    bf16* __restrict__ out) {
  int row = blockIdx.x;
  int tid = threadIdx.x;
  const float* xr = x + (size_t)row * HID;
  __shared__ float red[8];
  float v0 = xr[tid];
  float v1 = xr[tid + 256];
  float s = v0 * v0 + v1 * v1;
#pragma unroll
  for (int off = 16; off > 0; off >>= 1) s += __shfl_down(s, off, 32);
  if ((tid & 31) == 0) red[tid >> 5] = s;
  __syncthreads();
  if (tid < 32) {
    float t = (tid < 8) ? red[tid] : 0.f;
#pragma unroll
    for (int off = 4; off > 0; off >>= 1) t += __shfl_down(t, off, 32);
    if (tid == 0) red[0] = t;
  }
  __syncthreads();
  float rinv = rsqrtf(red[0] / (float)HID + EPSN);
  out[(size_t)row * HID + tid] = (bf16)(v0 * rinv * w[tid]);
  out[(size_t)row * HID + tid + 256] = (bf16)(v1 * rinv * w[tid + 256]);
}

// dt columns (2304..2308) of the input projection: 4 dots of length 512 / row
__global__ __launch_bounds__(128) void dt_kernel(
    const bf16* __restrict__ h, const float* __restrict__ w_in,
    float* __restrict__ dt_raw) {
  int row = blockIdx.x;
  int head = threadIdx.x >> 5;
  int lane = threadIdx.x & 31;
  const bf16* hr = h + (size_t)row * HID;
  float s = 0.f;
  for (int k = lane; k < HID; k += 32)
    s += (float)hr[k] * w_in[(size_t)k * PROJ_FULL + PROJ_W + head];
#pragma unroll
  for (int off = 16; off > 0; off >>= 1) s += __shfl_down(s, off, 32);
  if (lane == 0) dt_raw[(size_t)row * NH + head] = s;
}

// causal depthwise conv (K=4) + bias + SiLU on the xBC slice of zxbcdt
__global__ __launch_bounds__(256) void conv_silu_kernel(
    const float* __restrict__ zx, const float* __restrict__ cw,
    const float* __restrict__ cb, float* __restrict__ xout) {
  size_t idx = (size_t)blockIdx.x * 256 + threadIdx.x;
  int c = (int)(idx % CONVD);
  int row = (int)(idx / CONVD);
  int t = row % TLEN;
  int rbase = row - t;
  float s = cb[c];
#pragma unroll
  for (int k = 0; k < KCONV; ++k) {
    int tt = t - (KCONV - 1) + k;
    if (tt >= 0)
      s += zx[(size_t)(rbase + tt) * PROJ_W + INTER + c] * cw[c * KCONV + k];
  }
  xout[idx] = s / (1.f + __expf(-s));
}

// gate (y * silu(z)) + rmsnorm -> bf16
__global__ __launch_bounds__(256) void gate_norm_kernel(
    const float* __restrict__ y, const float* __restrict__ zx,
    const float* __restrict__ gw, bf16* __restrict__ out) {
  int row = blockIdx.x;
  int tid = threadIdx.x;
  const float* yr = y + (size_t)row * INTER;
  const float* zr = zx + (size_t)row * PROJ_W;  // z = cols [0,1024)
  __shared__ float red[8];
  float g[4];
  float s = 0.f;
#pragma unroll
  for (int i = 0; i < 4; ++i) {
    int c = tid + i * 256;
    float z = zr[c];
    float v = yr[c] * (z / (1.f + __expf(-z)));
    g[i] = v;
    s += v * v;
  }
#pragma unroll
  for (int off = 16; off > 0; off >>= 1) s += __shfl_down(s, off, 32);
  if ((tid & 31) == 0) red[tid >> 5] = s;
  __syncthreads();
  if (tid < 32) {
    float t = (tid < 8) ? red[tid] : 0.f;
#pragma unroll
    for (int off = 4; off > 0; off >>= 1) t += __shfl_down(t, off, 32);
    if (tid == 0) red[0] = t;
  }
  __syncthreads();
  float rinv = rsqrtf(red[0] / (float)INTER + EPSN);
#pragma unroll
  for (int i = 0; i < 4; ++i) {
    int c = tid + i * 256;
    out[(size_t)row * INTER + c] = (bf16)(g[i] * rinv * gw[c]);
  }
}

// ---------------- dense bf16 WMMA GEMM: C[M,N](f32) = A[M,K] @ B[K,N] --------
// 128x128 tile, BK=32, 256 threads = 8 waves; each wave owns a 64x32 sub-tile.
// Tiles staged via async global->LDS DMA when available.

template <bool ADD_RES>
__global__ __launch_bounds__(256) void gemm_bf16_kernel(
    const bf16* __restrict__ A, const bf16* __restrict__ B,
    const float* __restrict__ Res, float* __restrict__ C, int M, int N, int K) {
  __shared__ bf16 As[128][48];   // padded: row = 96B (16B aligned)
  __shared__ bf16 Bs[32][136];   // padded: row = 272B (16B aligned)
  const int tid = threadIdx.x;
  const int lane = tid & 31;
  const int wave = tid >> 5;
  const int wm = wave >> 2;  // 0..1 (rows)
  const int wn = wave & 3;   // 0..3 (cols)
  const int tile_n = blockIdx.x * 128;
  const int tile_m = blockIdx.y * 128;

  v8f acc[4][2];
#pragma unroll
  for (int i = 0; i < 4; ++i)
#pragma unroll
    for (int j = 0; j < 2; ++j) acc[i][j] = zero8();

  for (int k0 = 0; k0 < K; k0 += 32) {
#pragma unroll
    for (int i = 0; i < 2; ++i) {
      int q = tid * 2 + i;
      int r = q >> 2;
      int c = (q & 3) << 3;
      const bf16* src = A + (size_t)(tile_m + r) * K + k0 + c;
#ifdef USE_ASYNC_LDS
      async_cp16(&As[r][c], src);
#else
      *(uint4*)&As[r][c] = *(const uint4*)src;
#endif
    }
#pragma unroll
    for (int i = 0; i < 2; ++i) {
      int q = tid * 2 + i;
      int r = q >> 4;
      int c = (q & 15) << 3;
      const bf16* src = B + (size_t)(k0 + r) * N + tile_n + c;
#ifdef USE_ASYNC_LDS
      async_cp16(&Bs[r][c], src);
#else
      *(uint4*)&Bs[r][c] = *(const uint4*)src;
#endif
    }
    if (k0 + 32 < K) {  // prefetch next K slab into cache
      __builtin_prefetch(A + (size_t)(tile_m + (tid >> 1)) * K + k0 + 32, 0, 1);
      __builtin_prefetch(
          B + (size_t)(k0 + 32 + (tid >> 4)) * N + tile_n + ((tid & 15) << 3), 0, 1);
    }
#ifdef USE_ASYNC_LDS
    WAIT_ASYNC();  // s_barrier does not wait on ASYNCcnt
#endif
    __syncthreads();
    v16bf bfr[2];
#pragma unroll
    for (int nf = 0; nf < 2; ++nf)
      bfr[nf] = frag_b_bf(&Bs[0][0], 136, 0, wn * 32 + nf * 16, lane);
#pragma unroll
    for (int mf = 0; mf < 4; ++mf) {
      v16bf afr = frag_a_bf(&As[0][0], 48, wm * 64 + mf * 16, 0, lane);
#pragma unroll
      for (int nf = 0; nf < 2; ++nf)
        acc[mf][nf] = wmma_bf16(afr, bfr[nf], acc[mf][nf]);
    }
    __syncthreads();
  }

#pragma unroll
  for (int mf = 0; mf < 4; ++mf) {
#pragma unroll
    for (int nf = 0; nf < 2; ++nf) {
      int col = tile_n + wn * 32 + nf * 16 + (lane & 15);
      int row0 = tile_m + wm * 64 + mf * 16 + ((lane >> 4) << 3);
#pragma unroll
      for (int r = 0; r < 8; ++r) {
        size_t idx = (size_t)(row0 + r) * N + col;
        float v = acc[mf][nf][r];
        if (ADD_RES) v += Res[idx];
        C[idx] = v;
      }
    }
  }
}

// ---------------- chunked SSD scan (the sequential SSM recurrence) ----------
// One block per (b, h, p-tile of 32). State S[32][128] lives in LDS and is
// carried across 64 chunks of length 64. All per-chunk matmuls are WMMA.

__global__ __launch_bounds__(256) void scan_kernel(
    const float* __restrict__ xconv, const float* __restrict__ dt_raw,
    const float* __restrict__ dt_bias, const float* __restrict__ A_log,
    const float* __restrict__ Dp, float* __restrict__ y) {
  __shared__ float S[PT][NDIM];        // state, fp32 (16 KB)
  __shared__ float Xc[CHUNK][PT + 1];  // x chunk (this p-tile), fp32
  __shared__ bf16 Xw[PT][CHUNK + 8];   // decay-weighted x^T, bf16 (A of update)
  __shared__ bf16 Bc[CHUNK][136];      // B chunk, bf16 (padded)
  __shared__ bf16 Gm[CHUNK][72];       // masked C.B^T, bf16 (padded)
  __shared__ float dtv_l[CHUNK], cums_l[CHUNK], eT[CHUNK], wT[CHUNK];
  __shared__ float decay_tot_s;

  const int bid = blockIdx.x;   // 128 blocks
  const int ptile = bid & 7;
  const int h = (bid >> 3) & 3;
  const int b = bid >> 5;
  const int p0 = ptile * PT;
  const int tid = threadIdx.x;
  const int lane = tid & 31;
  const int wave = tid >> 5;

  const float A_h = -__expf(A_log[h]);
  const float D_h = Dp[h];
  const float dtb = dt_bias[h];
  const float* xb = xconv + (size_t)b * TLEN * CONVD;

  for (int i = tid; i < PT * NDIM; i += 256) ((float*)S)[i] = 0.f;
  __syncthreads();

  for (int c0 = 0; c0 < TLEN; c0 += CHUNK) {
    // ---- phase 1: stage chunk data, dt raw values ----
    for (int i = tid; i < CHUNK * PT; i += 256) {
      int t = i >> 5, p = i & (PT - 1);
      Xc[t][p] = xb[(size_t)(c0 + t) * CONVD + h * PDIM + p0 + p];
    }
    for (int i = tid; i < CHUNK * NDIM; i += 256) {
      int t = i >> 7, n = i & (NDIM - 1);
      Bc[t][n] = (bf16)xb[(size_t)(c0 + t) * CONVD + INTER + n];
    }
    if (tid < CHUNK) {
      float raw = dt_raw[(size_t)(b * TLEN + c0 + tid) * NH + h] + dtb;
      float dtv = (raw > 20.f) ? raw : log1pf(__expf(raw));
      dtv_l[tid] = dtv;
      wT[tid] = dtv * A_h;  // temp: per-step log decay
    }
    __syncthreads();
    // ---- phase 2: inclusive prefix of log-decay ----
    if (tid < CHUNK) {
      float s = 0.f;
      for (int i = 0; i <= tid; ++i) s += wT[i];
      cums_l[tid] = s;
    }
    __syncthreads();
    // ---- phase 3: decay weights ----
    if (tid < CHUNK) {
      float total = cums_l[CHUNK - 1];
      eT[tid] = __expf(cums_l[tid]);
      wT[tid] = dtv_l[tid] * __expf(total - cums_l[tid]);
      if (tid == 0) decay_tot_s = __expf(total);
    }
    __syncthreads();

    // ---- phase 3.5: build Xw^T once (shared by all waves in phase 6) ----
    for (int i = tid; i < CHUNK * PT; i += 256) {
      int p = i >> 6, t = i & (CHUNK - 1);
      Xw[p][t] = (bf16)(Xc[t][p] * wT[t]);
    }

    // ---- phase 4: G = C @ B^T (64x64, K=128); 2 output frags per wave ----
    {
      const int gf0 = wave * 2;
      const int mr = gf0 >> 2;  // both frags share the m-tile
      v8f gacc[2] = {zero8(), zero8()};
      const float* crow_base =
          xb + INTER + NDIM + (size_t)(c0 + mr * 16 + (lane & 15)) * CONVD;
#pragma unroll
      for (int ks = 0; ks < 4; ++ks) {
        int kn = ks * 32;
        const float* crow = crow_base + kn + ((lane >> 4) << 4);
        v16bf afr;
#pragma unroll
        for (int j = 0; j < 8; ++j) afr[j] = (bf16)crow[j];
#pragma unroll
        for (int j = 0; j < 8; ++j) afr[j + 8] = (bf16)crow[j + 8];
#pragma unroll
        for (int i = 0; i < 2; ++i) {
          int nc = (gf0 + i) & 3;
          v16bf bfr = frag_bT_bf(&Bc[0][0], 136, kn, nc * 16, lane);
          gacc[i] = wmma_bf16(afr, bfr, gacc[i]);
        }
      }
      // causal decay mask -> Gm (bf16)
#pragma unroll
      for (int i = 0; i < 2; ++i) {
        int nc = (gf0 + i) & 3;
        int s = nc * 16 + (lane & 15);
        int t0r = mr * 16 + ((lane >> 4) << 3);
#pragma unroll
        for (int r = 0; r < 8; ++r) {
          int t = t0r + r;
          float v = (s <= t)
                        ? gacc[i][r] * __expf(cums_l[t] - cums_l[s]) * dtv_l[s]
                        : 0.f;
          Gm[t][s] = (bf16)v;
        }
      }
    }
    __syncthreads();

    // ---- phase 5: Y = G' @ X  +  (e^cums * C) @ S^T  + D*x ; 1 frag/wave ----
    {
      const int ymr = wave >> 1;  // t tile 0..3
      const int ync = wave & 1;   // p tile 0..1
      v8f yacc = zero8();
#pragma unroll
      for (int ks = 0; ks < 2; ++ks) {  // intra-chunk, K=64
        v16bf afr = frag_a_bf(&Gm[0][0], 72, ymr * 16, ks * 32, lane);
        v16bf bfr = frag_b_f32(&Xc[0][0], PT + 1, ks * 32, ync * 16, lane);
        yacc = wmma_bf16(afr, bfr, yacc);
      }
      {
        const int m = ymr * 16 + (lane & 15);
        const float em = eT[m];
        const float* crow_base = xb + INTER + NDIM + (size_t)(c0 + m) * CONVD;
        const float* srow = &S[ync * 16 + (lane & 15)][0];
#pragma unroll
        for (int ks = 0; ks < 4; ++ks) {  // inter-chunk, K=128
          const float* crow = crow_base + ks * 32 + ((lane >> 4) << 4);
          v16bf afr;
#pragma unroll
          for (int j = 0; j < 8; ++j) afr[j] = (bf16)(crow[j] * em);
#pragma unroll
          for (int j = 0; j < 8; ++j) afr[j + 8] = (bf16)(crow[j + 8] * em);
          const float* sp = srow + ks * 32 + ((lane >> 4) << 4);
          v16bf bfr;
#pragma unroll
          for (int j = 0; j < 16; ++j) bfr[j] = (bf16)sp[j];
          yacc = wmma_bf16(afr, bfr, yacc);
        }
      }
      {
        int p = ync * 16 + (lane & 15);
        int t0r = ymr * 16 + ((lane >> 4) << 3);
        float* yb = y + (size_t)(b * TLEN + c0) * INTER + h * PDIM + p0;
#pragma unroll
        for (int r = 0; r < 8; ++r) {
          int t = t0r + r;
          yb[(size_t)t * INTER + p] = yacc[r] + D_h * Xc[t][p];
        }
      }
    }
    __syncthreads();  // all reads of S complete before update

    // ---- phase 6: S = decay_tot * S + Xw @ B  (32x128, K=64); 2 frags/wave --
    {
      const float dec = decay_tot_s;
      const int ncn = wave;  // n tile 0..7
      const int ncol = ncn * 16 + (lane & 15);
#pragma unroll
      for (int mr = 0; mr < 2; ++mr) {
        const int prow0 = mr * 16 + ((lane >> 4) << 3);
        v8f sacc;
#pragma unroll
        for (int r = 0; r < 8; ++r) sacc[r] = S[prow0 + r][ncol] * dec;
#pragma unroll
        for (int ks = 0; ks < 2; ++ks) {
          v16bf afr = frag_a_bf(&Xw[0][0], CHUNK + 8, mr * 16, ks * 32, lane);
          v16bf bfr = frag_b_bf(&Bc[0][0], 136, ks * 32, ncn * 16, lane);
          sacc = wmma_bf16(afr, bfr, sacc);
        }
#pragma unroll
        for (int r = 0; r < 8; ++r) S[prow0 + r][ncol] = sacc[r];
      }
    }
    __syncthreads();
  }
}

// ---------------------------------------------------------------------------

extern "C" void kernel_launch(void* const* d_in, const int* in_sizes, int n_in,
                              void* d_out, int out_size, void* d_ws,
                              size_t ws_size, hipStream_t stream) {
  const float* x      = (const float*)d_in[0];
  const float* w_in   = (const float*)d_in[1];
  const float* conv_w = (const float*)d_in[2];
  const float* conv_b = (const float*)d_in[3];
  const float* dt_b   = (const float*)d_in[4];
  const float* A_log  = (const float*)d_in[5];
  const float* Dv     = (const float*)d_in[6];
  const float* gnw    = (const float*)d_in[7];
  const float* w_out  = (const float*)d_in[8];
  const float* pnw    = (const float*)d_in[9];
  float* out = (float*)d_out;

  char* ws = (char*)d_ws;
  size_t off = 0;
  bf16* h_bf = (bf16*)(ws + off);  off += (size_t)ROWS * HID * 2;          // 16 MB
  bf16* wi_bf = (bf16*)(ws + off); off += (size_t)HID * PROJ_W * 2;        // 2.25 MB
  bf16* wo_bf = (bf16*)(ws + off); off += (size_t)INTER * HID * 2;         // 1 MB
  float* zx = (float*)(ws + off);  off += (size_t)ROWS * PROJ_W * 4;       // 151 MB
  float* dt_raw = (float*)(ws + off); off += (size_t)ROWS * NH * 4;        // 0.25 MB
  float* xconv = (float*)(ws + off);  off += (size_t)ROWS * CONVD * 4;     // 84 MB
  float* ybuf = (float*)(ws + off);   off += (size_t)ROWS * INTER * 4;     // 67 MB
  bf16* ygn = (bf16*)(ws + off);      off += (size_t)ROWS * INTER * 2;     // 32 MB

  cvt_w_in_kernel<<<(HID * PROJ_W) / 256, 256, 0, stream>>>(w_in, wi_bf);
  cvt_w_out_kernel<<<(INTER * HID) / 256, 256, 0, stream>>>(w_out, wo_bf);
  rmsnorm_in_kernel<<<ROWS, 256, 0, stream>>>(x, pnw, h_bf);

  // zxbcdt (z | xBC) = h @ w_in[:, :2304]
  gemm_bf16_kernel<false><<<dim3(PROJ_W / 128, ROWS / 128), 256, 0, stream>>>(
      h_bf, wi_bf, nullptr, zx, ROWS, PROJ_W, HID);

  dt_kernel<<<ROWS, 128, 0, stream>>>(h_bf, w_in, dt_raw);

  conv_silu_kernel<<<((size_t)ROWS * CONVD) / 256, 256, 0, stream>>>(
      zx, conv_w, conv_b, xconv);

  scan_kernel<<<BATCH * NH * (PDIM / PT), 256, 0, stream>>>(
      xconv, dt_raw, dt_b, A_log, Dv, ybuf);

  gate_norm_kernel<<<ROWS, 256, 0, stream>>>(ybuf, zx, gnw, ygn);

  // out = x + ygn @ w_out
  gemm_bf16_kernel<true><<<dim3(HID / 128, ROWS / 128), 256, 0, stream>>>(
      ygn, wo_bf, x, out, ROWS, HID, INTER);
}
